// MLPEncoder_13529146982623
// MI455X (gfx1250) — compile-verified, matrix operand order checked by reading, not run on hardware
//
#include <hip/hip_runtime.h>

typedef _Float16 half8 __attribute__((ext_vector_type(8)));
typedef __attribute__((ext_vector_type(16))) _Float16 v16h;
typedef __attribute__((ext_vector_type(8)))  float    v8f;

#define BB 16
#define NN 64
#define EE 4032
#define HH 4
#define DD 128
#define EPN 63

// ---- 16x32 f16 tile load per ISA 7.12.2 (A layout; B from row-major W[N,K] is
// the mirrored layout: lane holds one row of W). Two 16B chunks: K=8g..8g+7 and
// K=16+8g..16+8g+7. ----
__device__ __forceinline__ v16h ld16(const _Float16* p, int g) {
    half8 lo = *(const half8*)(p + 8 * g);
    half8 hi = *(const half8*)(p + 16 + 8 * g);
    v16h r;
#pragma unroll
    for (int i = 0; i < 8; i++) { r[i] = lo[i]; r[8 + i] = hi[i]; }
    return r;
}

// ---- edge index extraction from one-hot rel_rec / rel_send ----
__global__ void k_indices(const float* __restrict__ rec, const float* __restrict__ snd,
                          int* __restrict__ ridx, int* __restrict__ sidx) {
    int e = blockIdx.x * 256 + threadIdx.x;
    if (e >= EE) return;
    int r = 0, s = 0;
    for (int n = 0; n < NN; n++) {
        if (rec[e * NN + n] > 0.5f) r = n;
        if (snd[e * NN + n] > 0.5f) s = n;
    }
    ridx[e] = r; sidx[e] = s;
}

// ---- generic f32 -> f16 convert with K padding / column slicing / scale ----
__global__ void k_cvt(const float* __restrict__ src, _Float16* __restrict__ dst,
                      int rows, int srcld, int cols, int dstld, int coloff, float scale) {
    int i = blockIdx.x * 256 + threadIdx.x;
    int total = rows * dstld;
    if (i >= total) return;
    int r = i / dstld, c = i - r * dstld;
    float v = (c < cols) ? src[(size_t)r * srcld + coloff + c] * scale : 0.f;
    dst[i] = (_Float16)v;
}

// ---- generic WMMA GEMM: C = act(A[M,K](f16) @ W[N,K]^T(f16) + bias) ----
// one wave per 16x16 tile, 8 waves/block cover 128 output cols; optional A-row
// gather (row r -> A row b*gN + gidx[r%gE], b=r/gE) used for node2edge.
__global__ __launch_bounds__(256) void k_gemm(
    const _Float16* __restrict__ A, int lda,
    const _Float16* __restrict__ W, int K,
    const float* __restrict__ bias,
    float* __restrict__ oF32, _Float16* __restrict__ oF16, int ldo,
    int Nout, int act,
    const int* __restrict__ gidx, int gE, int gN) {
    int tid = threadIdx.x, wave = tid >> 5, lane = tid & 31, g = lane >> 4, lm = lane & 15;
    int ncol0 = (blockIdx.y * 8 + wave) * 16;
    if (ncol0 >= Nout) return;
    int m0 = blockIdx.x * 16;
    int r = m0 + lm;
    int ar;
    if (gE > 0) { int b = r / gE; int e = r - b * gE; ar = b * gN + gidx[e]; }
    else ar = r;
    const _Float16* Ap = A + (size_t)ar * lda;
    const _Float16* Wp = W + (size_t)(ncol0 + lm) * K;
    v8f c = {};
    int nk = K >> 5;
    for (int kt = 0; kt < nk; kt++) {
        v16h a = ld16(Ap + kt * 32, g);
        v16h w = ld16(Wp + kt * 32, g);
        c = __builtin_amdgcn_wmma_f32_16x16x32_f16(false, a, false, w, (short)0, c, false, false);
    }
    int col = ncol0 + lm;
    float bv = bias ? bias[col] : 0.f;
#pragma unroll
    for (int v = 0; v < 8; v++) {
        int row = m0 + v + 8 * g;
        float x = c[v] + bv;
        if (act == 1) x = x > 0.f ? x : (expf(x) - 1.f);   // ELU
        else if (act == 2) x = x > 0.f ? x : 0.f;          // ReLU
        if (oF32) oF32[(size_t)row * ldo + col] = x;
        if (oF16) oF16[(size_t)row * ldo + col] = (_Float16)x;
    }
}

// ---- per-node attention dots: attn = leaky_relu(v_proj . a)/500 ----
__global__ void k_nodeattn(const _Float16* __restrict__ vproj, const float* __restrict__ afwd,
                           const float* __restrict__ abwd, float* __restrict__ attF,
                           float* __restrict__ attB) {
    int i = blockIdx.x * 256 + threadIdx.x;
    if (i >= BB * NN * HH) return;
    int h = i & 3;
    const _Float16* p = vproj + (size_t)i * DD;
    float df = 0.f, db = 0.f;
    for (int k = 0; k < DD; k++) {
        float x = (float)p[k];
        df = fmaf(x, afwd[h * DD + k], df);
        db = fmaf(x, abwd[h * DD + k], db);
    }
    attF[i] = (df > 0.f ? df : 0.2f * df) * (1.f / 500.f);
    attB[i] = (db > 0.f ? db : 0.2f * db) * (1.f / 500.f);
}

// ---- per-edge alpha (faithful double-exp) ----
__global__ void k_alpha(const float* __restrict__ attF, const float* __restrict__ attB,
                        const int* __restrict__ ridx, const int* __restrict__ sidx,
                        float* __restrict__ aij, float* __restrict__ aji) {
    int i = blockIdx.x * 256 + threadIdx.x;
    if (i >= BB * EE * HH) return;
    int h = i & 3; int be = i >> 2; int e = be % EE; int b = be / EE;
    int s = sidx[e], rr = ridx[e];
    float aI = attF[((b * NN + s) << 2) + h];
    float aJ = attB[((b * NN + rr) << 2) + h];
    float m = fmaxf(aI, aJ);
    float eI = expf(expf(aI - m));
    float eJ = expf(expf(aJ - m));
    float den = eI + eJ;
    aij[i] = eI / den;
    aji[i] = eJ / den;
}

// ---- fused edge layer-2 GEMM + alpha-weighted scatter to v_social ----
// one block per (b,n): 252 edge-rows (63 edges x 4 heads) contiguous by
// construction. Build relu(a_ij*P1[src] + a_ji*P2[dst] + b1) f16 in LDS,
// WMMA vs e_w2, reduce alpha_ij-weighted rows per head in registers.
__global__ __launch_bounds__(256) void k_edge(
    const float* __restrict__ P1, const float* __restrict__ P2,
    const float* __restrict__ aij, const float* __restrict__ aji,
    const float* __restrict__ eb1, const _Float16* __restrict__ W2,
    const int* __restrict__ sidx, _Float16* __restrict__ vsoc) {
    int bn = blockIdx.x; int b = bn >> 6; int n = bn & 63;
    __shared__ _Float16 At[128 * 136];   // 34.8KB, padded stride: bank-conflict-free
    __shared__ float s_aij[256];
    __shared__ float s_aji[256];
    int tid = threadIdx.x;
    {
        int q = tid >> 2, h = tid & 3;
        float x = 0.f, y = 0.f;
        if (q < EPN) {
            int e = n * EPN + q;
            x = aij[((size_t)(b * EE + e) << 2) + h];
            y = aji[((size_t)(b * EE + e) << 2) + h];
        }
        s_aij[tid] = x; s_aji[tid] = y;
    }
    int wave = tid >> 5, lane = tid & 31, g = lane >> 4, lm = lane & 15;
    v16h Bt[4];
#pragma unroll
    for (int kt = 0; kt < 4; kt++)
        Bt[kt] = ld16(W2 + (size_t)(wave * 16 + lm) * DD + kt * 32, g);
    float acc0 = 0.f, acc1 = 0.f, acc2 = 0.f, acc3 = 0.f;
    __syncthreads();
    for (int chunk = 0; chunk < 2; chunk++) {
        for (int idx = tid; idx < 128 * 128; idx += 256) {
            int rl = idx >> 7, k = idx & 127;
            int r = chunk * 128 + rl;
            float v = 0.f;
            if (r < EPN * HH) {
                int q = r >> 2, h = r & 3, e = n * EPN + q, s = sidx[e];
                float p1 = P1[(size_t)(((b * NN + s) << 2) + h) * DD + k];
                float p2 = P2[(size_t)(((b * NN + n) << 2) + h) * DD + k];
                v = fmaf(s_aij[r], p1, fmaf(s_aji[r], p2, eb1[k]));
                v = v > 0.f ? v : 0.f;   // ReLU
            }
            At[rl * 136 + k] = (_Float16)v;
        }
        __syncthreads();
        for (int t = 0; t < 8; t++) {
            v8f c = {};
#pragma unroll
            for (int kt = 0; kt < 4; kt++) {
                v16h a = ld16(&At[(t * 16 + lm) * 136 + kt * 32], g);
                c = __builtin_amdgcn_wmma_f32_16x16x32_f16(false, a, false, Bt[kt], (short)0, c, false, false);
            }
            // C layout: c[v] is row M=v+8g, col=lm; head h = row%4 = v&3,
            // edge q = row/4 = chunk*32 + 4t + 2g + (v>>2). padded rows have alpha 0.
            int qb = chunk * 32 + 4 * t + 2 * g;
#pragma unroll
            for (int v = 0; v < 8; v++) {
                float al = s_aij[((qb + (v >> 2)) << 2) + (v & 3)];
                float cv = c[v];
                if ((v & 3) == 0) acc0 = fmaf(al, cv, acc0);
                else if ((v & 3) == 1) acc1 = fmaf(al, cv, acc1);
                else if ((v & 3) == 2) acc2 = fmaf(al, cv, acc2);
                else acc3 = fmaf(al, cv, acc3);
            }
        }
        __syncthreads();
    }
    acc0 += __shfl_xor(acc0, 16, 32);
    acc1 += __shfl_xor(acc1, 16, 32);
    acc2 += __shfl_xor(acc2, 16, 32);
    acc3 += __shfl_xor(acc3, 16, 32);
    if (lane < 16) {
        _Float16* o = vsoc + (size_t)(b * NN + n) * (HH * DD) + wave * 16 + lane;
        o[0] = (_Float16)acc0; o[DD] = (_Float16)acc1;
        o[2 * DD] = (_Float16)acc2; o[3 * DD] = (_Float16)acc3;
    }
}

// ---- edge2node segmented sums (recv contiguous, send analytic) + concat ----
__global__ void k_e2n_cat(const _Float16* __restrict__ xe3, const _Float16* __restrict__ xx,
                          _Float16* __restrict__ cat) {
    int i = blockIdx.x * 256 + threadIdx.x;
    if (i >= BB * NN * 128) return;
    int d = i & 127; int bn = i >> 7; int n = bn & 63; int b = bn >> 6;
    float rs = 0.f, ss = 0.f;
    const _Float16* base = xe3 + (size_t)b * EE * 128 + d;
    for (int q = 0; q < EPN; q++) rs += (float)base[(size_t)(n * EPN + q) * 128];
    for (int i2 = 0; i2 < NN; i2++) {
        if (i2 == n) continue;
        int e = i2 * EPN + n - (n > i2 ? 1 : 0);
        ss += (float)base[(size_t)e * 128];
    }
    _Float16* o = cat + (size_t)bn * 384;
    o[d] = (_Float16)ss;
    o[128 + d] = (_Float16)rs;
    o[256 + d] = xx[(size_t)bn * 128 + d];
}

extern "C" void kernel_launch(void* const* d_in, const int* in_sizes, int n_in,
                              void* d_out, int out_size, void* d_ws, size_t ws_size,
                              hipStream_t stream) {
    const float* inp     = (const float*)d_in[0];
    const float* rel_rec = (const float*)d_in[1];
    const float* rel_snd = (const float*)d_in[2];
    const float* fh_w1 = (const float*)d_in[3];  const float* fh_b1 = (const float*)d_in[4];
    const float* fh_w2 = (const float*)d_in[5];  const float* fh_b2 = (const float*)d_in[6];
    const float* fh_w3 = (const float*)d_in[7];  const float* fh_b3 = (const float*)d_in[8];
    const float* proj_w = (const float*)d_in[9];
    const float* a_fwd = (const float*)d_in[10]; const float* a_bwd = (const float*)d_in[11];
    const float* e_w1 = (const float*)d_in[12];  const float* e_b1 = (const float*)d_in[13];
    const float* e_w2 = (const float*)d_in[14];  const float* e_b2 = (const float*)d_in[15];
    const float* cg_w1 = (const float*)d_in[16]; const float* cg_b1 = (const float*)d_in[17];
    const float* cg_w2 = (const float*)d_in[18]; const float* cg_b2 = (const float*)d_in[19];
    const float* cg_w3 = (const float*)d_in[20]; const float* cg_b3 = (const float*)d_in[21];
    const float* m3_w1 = (const float*)d_in[22]; const float* m3_b1 = (const float*)d_in[23];
    const float* m3_w2 = (const float*)d_in[24]; const float* m3_b2 = (const float*)d_in[25];
    const float* m3_w3 = (const float*)d_in[26]; const float* m3_b3 = (const float*)d_in[27];
    const float* m4_w1 = (const float*)d_in[28]; const float* m4_b1 = (const float*)d_in[29];
    const float* m4_w2 = (const float*)d_in[30]; const float* m4_b2 = (const float*)d_in[31];
    const float* m4_w3 = (const float*)d_in[32]; const float* m4_b3 = (const float*)d_in[33];
    const float* out_w = (const float*)d_in[34]; const float* out_b = (const float*)d_in[35];

    float* out_main  = (float*)d_out;                 // [16,64,64]
    float* out_vself = (float*)d_out + 65536;         // [16,64,128]
    float* out_aij   = (float*)d_out + 196608;        // [16,4032,4]

    // ---- workspace bump allocator ----
    char* wsp = (char*)d_ws; size_t off = 0;
    auto alloc = [&](size_t bytes) -> void* {
        void* p = wsp + off; off = (off + bytes + 255) & ~(size_t)255; return p;
    };
    int* ridx = (int*)alloc(EE * 4);
    int* sidx = (int*)alloc(EE * 4);
    _Float16* w_fh1 = (_Float16*)alloc(128 * 32 * 2);
    _Float16* w_fh2 = (_Float16*)alloc(128 * 128 * 2);
    _Float16* w_fh3 = (_Float16*)alloc(128 * 128 * 2);
    _Float16* w_proj = (_Float16*)alloc(512 * 128 * 2);
    _Float16* w_e1s = (_Float16*)alloc(128 * 128 * 2);
    _Float16* w_e1t = (_Float16*)alloc(128 * 128 * 2);
    _Float16* w_e2  = (_Float16*)alloc(128 * 128 * 2);
    _Float16* w_cg1 = (_Float16*)alloc(128 * 512 * 2);
    _Float16* w_cg2 = (_Float16*)alloc(128 * 128 * 2);
    _Float16* w_cg3 = (_Float16*)alloc(128 * 128 * 2);
    _Float16* w_m31 = (_Float16*)alloc(128 * 128 * 2);
    _Float16* w_m32 = (_Float16*)alloc(128 * 128 * 2);
    _Float16* w_m33 = (_Float16*)alloc(128 * 128 * 2);
    _Float16* w_m41 = (_Float16*)alloc(128 * 384 * 2);
    _Float16* w_m42 = (_Float16*)alloc(128 * 128 * 2);
    _Float16* w_m43 = (_Float16*)alloc(128 * 128 * 2);
    _Float16* w_out = (_Float16*)alloc(64 * 128 * 2);
    _Float16* xpad  = (_Float16*)alloc(1024 * 32 * 2);
    _Float16* hA    = (_Float16*)alloc(1024 * 128 * 2);
    _Float16* hB    = (_Float16*)alloc(1024 * 128 * 2);
    _Float16* vself16 = (_Float16*)alloc(1024 * 128 * 2);
    _Float16* vproj16 = (_Float16*)alloc(1024 * 512 * 2);
    float* attF  = (float*)alloc(4096 * 4);
    float* attB  = (float*)alloc(4096 * 4);
    float* aji   = (float*)alloc((size_t)BB * EE * HH * 4);
    float* P1    = (float*)alloc((size_t)4096 * 128 * 4);
    float* P2    = (float*)alloc((size_t)4096 * 128 * 4);
    _Float16* vsoc16 = (_Float16*)alloc(1024 * 512 * 2);
    _Float16* xx16   = (_Float16*)alloc(1024 * 128 * 2);
    _Float16* eA = (_Float16*)alloc((size_t)64512 * 128 * 2);
    _Float16* eB = (_Float16*)alloc((size_t)64512 * 128 * 2);
    _Float16* cat16 = (_Float16*)alloc(1024 * 384 * 2);
    _Float16* xn16  = (_Float16*)alloc(1024 * 128 * 2);
    (void)ws_size; (void)in_sizes; (void)n_in; (void)out_size;

    auto cvt = [&](const float* s, _Float16* dst, int rows, int srcld, int cols,
                   int dstld, int coloff, float scale) {
        int tot = rows * dstld;
        k_cvt<<<(tot + 255) / 256, 256, 0, stream>>>(s, dst, rows, srcld, cols, dstld, coloff, scale);
    };

    k_indices<<<(EE + 255) / 256, 256, 0, stream>>>(rel_rec, rel_snd, ridx, sidx);
    cvt(fh_w1, w_fh1, 128, 30, 30, 32, 0, 1.f);
    cvt(fh_w2, w_fh2, 128, 128, 128, 128, 0, 1.f);
    cvt(fh_w3, w_fh3, 128, 128, 128, 128, 0, 1.f);
    cvt(proj_w, w_proj, 512, 128, 128, 128, 0, 1.f);
    cvt(e_w1, w_e1s, 128, 256, 128, 128, 0, 1.f);     // first 128 cols -> src half
    cvt(e_w1, w_e1t, 128, 256, 128, 128, 128, 1.f);   // last  128 cols -> tgt half
    cvt(e_w2, w_e2, 128, 128, 128, 128, 0, 1.f);
    cvt(cg_w1, w_cg1, 128, 512, 512, 512, 0, 1.f);
    cvt(cg_w2, w_cg2, 128, 128, 128, 128, 0, 1.f);
    cvt(cg_w3, w_cg3, 128, 128, 128, 128, 0, 1.f);
    cvt(m3_w1, w_m31, 128, 128, 128, 128, 0, 1.f / (float)EE); // fold node2edge /E
    cvt(m3_w2, w_m32, 128, 128, 128, 128, 0, 1.f);
    cvt(m3_w3, w_m33, 128, 128, 128, 128, 0, 1.f);
    cvt(m4_w1, w_m41, 128, 384, 384, 384, 0, 1.f);
    cvt(m4_w2, w_m42, 128, 128, 128, 128, 0, 1.f);
    cvt(m4_w3, w_m43, 128, 128, 128, 128, 0, 1.f);
    cvt(out_w, w_out, 64, 128, 128, 128, 0, 1.f);
    cvt(inp, xpad, 1024, 30, 30, 32, 0, 1.f);         // [B,N,T*Fin] -> K padded to 32

    dim3 blk(256);
    auto gemm = [&](const _Float16* A, int lda, const _Float16* W, int K,
                    const float* bias, float* o32, _Float16* o16, int ldo,
                    int M, int Nout, int act, const int* gidx, int gE, int gN) {
        dim3 grid(M / 16, (Nout + 127) / 128);
        k_gemm<<<grid, blk, 0, stream>>>(A, lda, W, K, bias, o32, o16, ldo, Nout, act, gidx, gE, gN);
    };

    // v_self = MLP(x): ELU, ELU, linear
    gemm(xpad, 32, w_fh1, 32, fh_b1, nullptr, hA, 128, 1024, 128, 1, nullptr, 0, 0);
    gemm(hA, 128, w_fh2, 128, fh_b2, nullptr, hB, 128, 1024, 128, 1, nullptr, 0, 0);
    gemm(hB, 128, w_fh3, 128, fh_b3, out_vself, vself16, 128, 1024, 128, 0, nullptr, 0, 0);
    // v_proj = v_self @ proj_w^T  -> [B,N,H,D]
    gemm(vself16, 128, w_proj, 128, nullptr, nullptr, vproj16, 512, 1024, 512, 0, nullptr, 0, 0);
    // per-node attention scalars; per-edge alpha (output + ws)
    k_nodeattn<<<(4096 + 255) / 256, 256, 0, stream>>>(vproj16, a_fwd, a_bwd, attF, attB);
    k_alpha<<<((BB * EE * HH) + 255) / 256, 256, 0, stream>>>(attF, attB, ridx, sidx, out_aij, aji);
    // P1 = v_proj @ W1s^T, P2 = v_proj @ W1t^T (node-level, f32)
    gemm(vproj16, 128, w_e1s, 128, nullptr, P1, nullptr, 128, 4096, 128, 0, nullptr, 0, 0);
    gemm(vproj16, 128, w_e1t, 128, nullptr, P2, nullptr, 128, 4096, 128, 0, nullptr, 0, 0);
    // fused edge layer-2 + alpha-weighted scatter -> v_social
    k_edge<<<BB * NN, blk, 0, stream>>>(P1, P2, out_aij, aji, e_b1, w_e2, sidx, vsoc16);
    // xx = MLP_cg(v_social)
    gemm(vsoc16, 512, w_cg1, 512, cg_b1, nullptr, hA, 128, 1024, 128, 1, nullptr, 0, 0);
    gemm(hA, 128, w_cg2, 128, cg_b2, nullptr, hB, 128, 1024, 128, 1, nullptr, 0, 0);
    gemm(hB, 128, w_cg3, 128, cg_b3, nullptr, xx16, 128, 1024, 128, 0, nullptr, 0, 0);
    // m3 MLP on edges (gathered node2edge rows; /E folded into w_m31)
    gemm(xx16, 128, w_m31, 128, m3_b1, nullptr, eA, 128, 64512, 128, 1, ridx, EE, NN);
    gemm(eA, 128, w_m32, 128, m3_b2, nullptr, eB, 128, 64512, 128, 1, nullptr, 0, 0);
    gemm(eB, 128, w_m33, 128, m3_b3, nullptr, eA, 128, 64512, 128, 0, nullptr, 0, 0);
    // edge2node sums + concat [send, recv, x_skip]
    k_e2n_cat<<<((BB * NN * 128) + 255) / 256, 256, 0, stream>>>(eA, xx16, cat16);
    // m4 MLP + output projection
    gemm(cat16, 384, w_m41, 384, m4_b1, nullptr, hA, 128, 1024, 128, 1, nullptr, 0, 0);
    gemm(hA, 128, w_m42, 128, m4_b2, nullptr, hB, 128, 1024, 128, 1, nullptr, 0, 0);
    gemm(hB, 128, w_m43, 128, m4_b3, nullptr, xn16, 128, 1024, 128, 0, nullptr, 0, 0);
    gemm(xn16, 128, w_out, 128, out_b, out_main, nullptr, 64, 1024, 64, 0, nullptr, 0, 0);
}
